// NGPModel_9191230014086
// MI455X (gfx1250) — matrix operand order repaired and checked by reading, not compiled, not run on hardware
//
#include <hip/hip_runtime.h>

typedef __attribute__((ext_vector_type(16))) _Float16 v16h;
typedef __attribute__((ext_vector_type(2)))  _Float16 v2h;
typedef __attribute__((ext_vector_type(8)))  float    v8f;
typedef __attribute__((ext_vector_type(4)))  float    v4f;

#define N_LEVELS      16
#define T_SIZE        (1u << 19)         // 524288 entries per level
#define HASH_PRIME1   2654435761u
#define WAVES_PER_BLK 8
#define LDS_STRIDE    72                 // f16 per row: 64 used + 8 pad (144B, 16B aligned)
#define ROWS_PER_WAVE 32
#define LVL_GROUP     4                  // levels per gather batch (16 loads in flight)

__global__ __launch_bounds__(256) void ngp_fused_kernel(
    const float* __restrict__ pos,    // [N,2]
    const float* __restrict__ table,  // [16, T, 2]
    const float* __restrict__ W1,     // [32,64]
    const float* __restrict__ W2,     // [64,64]
    const float* __restrict__ W3,     // [64,3]
    float* __restrict__ out,          // [N,3]
    int nPoints)
{
    __shared__ __align__(16) _Float16 lds[WAVES_PER_BLK * ROWS_PER_WAVE * LDS_STRIDE];

    const int lane      = threadIdx.x & 31;
    const int waveInBlk = threadIdx.x >> 5;
    const int wave      = blockIdx.x * WAVES_PER_BLK + waveInBlk;
    const int nWaves    = gridDim.x * WAVES_PER_BLK;
    _Float16* myLds     = &lds[waveInBlk * ROWS_PER_WAVE * LDS_STRIDE];

    const int nLo = lane & 15;   // point (N) / feature-row (M) index inside fragments
    const int hiK = lane >> 4;   // 0: lanes 0-15, 1: lanes 16-31

    // ---- Weights live in A-fragments of W^T (f16), loaded once into VGPRs ----
    // Transposed orientation: D^T = W^T x h^T, so M = output features, N = points.
    // A layout (16-bit 16x32 MxK): lane<16 -> M=lane, K 0-7 & 16-23;
    //                              lane>=16 -> M=lane-16, K 8-15 & 24-31.
    // A[m][k] = W[k][m] (W stored row-major [Kin][Nout]).
    auto loadWA = [&](const float* __restrict__ W, int Nout, int mbase, int kbase) -> v16h {
        v16h a;
        const int m  = mbase + nLo;
        const int k0 = kbase + hiK * 8;
        #pragma unroll
        for (int i = 0; i < 8; ++i) {
            const float wlo = (m < Nout) ? W[(size_t)(k0 + i)      * (size_t)Nout + (size_t)m] : 0.0f;
            const float whi = (m < Nout) ? W[(size_t)(k0 + 16 + i) * (size_t)Nout + (size_t)m] : 0.0f;
            a[i]     = (_Float16)wlo;
            a[8 + i] = (_Float16)whi;
        }
        return a;
    };

    v16h A1[4], A2[8], A3[2];
    #pragma unroll
    for (int mt = 0; mt < 4; ++mt) A1[mt] = loadWA(W1, 64, mt * 16, 0);
    #pragma unroll
    for (int mt = 0; mt < 4; ++mt)
        #pragma unroll
        for (int c = 0; c < 2; ++c) A2[mt * 2 + c] = loadWA(W2, 64, mt * 16, c * 32);
    #pragma unroll
    for (int c = 0; c < 2; ++c) A3[c] = loadWA(W3, 3, 0, c * 32);

    // B-fragment (activations h^T, 32x16 KxN) from row-major [point][feature] LDS:
    // lane<16 -> col N=lane (point), K 0-7 & 16-23; lane>=16 -> same point, K 8-15 & 24-31.
    // Per lane: two 8-f16 runs of one point's features -> 2x ds_load_b128.
    auto loadBH = [&](int ptBase, int kbase) -> v16h {
        const int row = ptBase + nLo;           // point
        const int c0  = kbase + hiK * 8;        // feature base for this lane
        union { v16h h; v4f f[2]; } u;
        u.f[0] = *(const v4f*)&myLds[row * LDS_STRIDE + c0];
        u.f[1] = *(const v4f*)&myLds[row * LDS_STRIDE + c0 + 16];
        return u.h;
    };

    // D^T layout: VGPR r -> out-feature mt*16 + hiK*8 + r, lane -> point.
    // Each lane holds 8 consecutive features of one point. Convert first
    // (v_cvt_pk_f16_f32, known-canonical), then packed-f16 ReLU
    // (v_pk_max_num_f16, no canonicalize needed), then ONE ds_store_b128.
    auto storeH = [&](int ptBase, int mt, const v8f& d) {
        const v2h z = {(_Float16)0.0f, (_Float16)0.0f};
        union { v2h p[4]; v4f f; } u;
        #pragma unroll
        for (int j = 0; j < 4; ++j) {
            v2h p; p[0] = (_Float16)d[2 * j]; p[1] = (_Float16)d[2 * j + 1];
            u.p[j] = __builtin_elementwise_max(p, z);
        }
        const int row = ptBase + nLo;            // point
        const int col = mt * 16 + hiK * 8;       // feature base (16B aligned)
        *(v4f*)&myLds[row * LDS_STRIDE + col] = u.f;
    };

    for (int base = wave * 32; base < nPoints; base += nWaves * 32) {
        // gfx1250 prefetch of next tile's positions
        const int nb = base + nWaves * 32;
        if (nb < nPoints)
            __builtin_prefetch(&pos[(size_t)(nb + lane) * 2], 0, 0);

        // ================= hash-grid encode: one point per lane =================
        // Batched gathers: per 4-level group, compute all 16 corner indices and
        // issue all 16 8B gathers before consuming -> memory-level parallelism.
        const float2 xy = ((const float2*)pos)[base + lane];
        #pragma unroll
        for (int lg = 0; lg < N_LEVELS; lg += LVL_GROUP) {
            float2 fv[4 * LVL_GROUP];
            float  wt[4 * LVL_GROUP];
            #pragma unroll
            for (int l = 0; l < LVL_GROUP; ++l) {
                const int lvl = lg + l;
                const float scale = (float)(16u << lvl) - 1.0f;   // compile-time
                const unsigned res = 16u << lvl;
                const float px = fmaf(xy.x, scale, 0.5f);
                const float py = fmaf(xy.y, scale, 0.5f);
                const float fx = floorf(px), fy = floorf(py);
                const float wx1 = px - fx, wy1 = py - fy;
                const unsigned ix = (unsigned)fx, iy = (unsigned)fy;
                const float2* tl2 = (const float2*)table + (size_t)lvl * T_SIZE;
                #pragma unroll
                for (int corner = 0; corner < 4; ++corner) {
                    const unsigned cx = ix + (unsigned)(corner & 1);
                    const unsigned cy = iy + (unsigned)(corner >> 1);
                    unsigned idx = (lvl <= 5) ? (cx + cy * res)            // dense
                                              : (cx ^ (cy * HASH_PRIME1)); // hashed
                    idx &= (T_SIZE - 1u);
                    fv[l * 4 + corner] = tl2[idx];                          // 8B gather
                    wt[l * 4 + corner] = ((corner & 1) ? wx1 : (1.0f - wx1))
                                       * ((corner >> 1) ? wy1 : (1.0f - wy1));
                }
            }
            #pragma unroll
            for (int l = 0; l < LVL_GROUP; ++l) {
                float f0 = 0.0f, f1 = 0.0f;
                #pragma unroll
                for (int corner = 0; corner < 4; ++corner) {
                    f0 = fmaf(wt[l * 4 + corner], fv[l * 4 + corner].x, f0);
                    f1 = fmaf(wt[l * 4 + corner], fv[l * 4 + corner].y, f1);
                }
                v2h hv; hv[0] = (_Float16)f0; hv[1] = (_Float16)f1;
                *(v2h*)&myLds[lane * LDS_STRIDE + 2 * (lg + l)] = hv;  // ds_store_b32
            }
        }

        // ============ WMMA MLP (transposed): two 16-point groups per wave ============
        #pragma unroll
        for (int pt = 0; pt < 2; ++pt) {
            const int ptBase = pt * 16;

            // layer 1: W1^T[64x32] x x^T[32x16] -> 4 independent M-tiles, K=32.
            // Compute all 4 accumulators first so the WMMAs issue back-to-back
            // and ReLU/store of tile i covers the D-read hazard of tile i+1.
            {
                const v16h b = loadBH(ptBase, 0);
                v8f acc[4];
                #pragma unroll
                for (int mt = 0; mt < 4; ++mt) {
                    v8f c = {};
                    acc[mt] = __builtin_amdgcn_wmma_f32_16x16x32_f16(
                                  false, A1[mt], false, b, (short)0, c, false, false);
                }
                #pragma unroll
                for (int mt = 0; mt < 4; ++mt) storeH(ptBase, mt, acc[mt]);
            }

            // layer 2: W2^T[64x64] x h1^T[64x16] -> 4 M-tiles x 2 K-chunks
            {
                const v16h b0 = loadBH(ptBase, 0);
                const v16h b1 = loadBH(ptBase, 32);
                v8f acc[4];
                #pragma unroll
                for (int mt = 0; mt < 4; ++mt) {
                    v8f c = {};
                    c = __builtin_amdgcn_wmma_f32_16x16x32_f16(
                            false, A2[mt * 2 + 0], false, b0, (short)0, c, false, false);
                    acc[mt] = __builtin_amdgcn_wmma_f32_16x16x32_f16(
                            false, A2[mt * 2 + 1], false, b1, (short)0, c, false, false);
                }
                #pragma unroll
                for (int mt = 0; mt < 4; ++mt) storeH(ptBase, mt, acc[mt]);
            }

            // layer 3: W3^T[3(pad 16)x64] x h2^T[64x16]; no activation
            {
                const v16h b0 = loadBH(ptBase, 0);
                const v16h b1 = loadBH(ptBase, 32);
                v8f c = {};
                c = __builtin_amdgcn_wmma_f32_16x16x32_f16(
                        false, A3[0], false, b0, (short)0, c, false, false);
                c = __builtin_amdgcn_wmma_f32_16x16x32_f16(
                        false, A3[1], false, b1, (short)0, c, false, false);

                // lane n<16 holds out-features 0..2 of point ptBase+n in c[0..2]
                if (lane < 16) {
                    const int point = base + ptBase + nLo;
                    float* o = &out[(size_t)point * 3];
                    o[0] = c[0];
                    o[1] = c[1];
                    o[2] = c[2];
                }
            }
        }
    }
}

extern "C" void kernel_launch(void* const* d_in, const int* in_sizes, int n_in,
                              void* d_out, int out_size, void* d_ws, size_t ws_size,
                              hipStream_t stream) {
    const float* pos   = (const float*)d_in[0];  // [N,2]
    const float* table = (const float*)d_in[1];  // [16,T,2]
    const float* W1    = (const float*)d_in[2];  // [32,64]
    const float* W2    = (const float*)d_in[3];  // [64,64]
    const float* W3    = (const float*)d_in[4];  // [64,3]
    float* out = (float*)d_out;                  // [N,3]
    const int nPoints = in_sizes[0] / 2;         // 4194304

    const dim3 grid(1024);   // 8192 waves, grid-stride over 131072 32-point tiles
    const dim3 block(256);   // 8 waves/block; wave-private LDS slabs, no barriers
    hipLaunchKernelGGL(ngp_fused_kernel, grid, block, 0, stream,
                       pos, table, W1, W2, W3, out, nPoints);
}
